// LSTM_88708254531702
// MI455X (gfx1250) — matrix operand order repaired and checked by reading, not compiled, not run on hardware
//
#include <hip/hip_runtime.h>
#include <hip/hip_bf16.h>

// ---------------------------------------------------------------------------
// 3-layer LSTM on MI455X (gfx1250), bf16 WMMA + f32 accumulate.
// Per (t, layer): one kernel. Grid (H/16, B/16), 128 threads (4 waves).
// Wave g computes the 16x16 tile of gate g (i,f,g,o) at columns g*H + n0,
// shares via LDS, then the workgroup does the pointwise cell update.
// Weights are converted to bf16 once into workspace (fit in L2 = 192MB),
// h-state kept in bf16 double buffer (t parity) for WMMA A-operand reuse,
// c-state and final h live in f32 directly inside d_out.
// ---------------------------------------------------------------------------

typedef unsigned short u16;
typedef __attribute__((ext_vector_type(16))) __bf16 v16bf;
typedef __attribute__((ext_vector_type(8)))  __bf16 v8bf;
typedef __attribute__((ext_vector_type(8)))  float  v8f;

#define T_STEPS 512
#define BATCH   64
#define DIN     256
#define HID     512
#define NLAYER  3
#define G4H     2048   // 4*HID

__device__ __forceinline__ u16 f32_to_bf16_rne(float f) {
  unsigned int u = __float_as_uint(f);
  u = u + 0x7FFFu + ((u >> 16) & 1u);   // round-to-nearest-even
  return (u16)(u >> 16);
}

__global__ void k_cvt_bf16(const float* __restrict__ src, u16* __restrict__ dst, int n) {
  int i = blockIdx.x * blockDim.x + threadIdx.x;
  if (i < n) dst[i] = f32_to_bf16_rne(src[i]);
}

__global__ void k_add_f32(const float* __restrict__ a, const float* __restrict__ b,
                          float* __restrict__ o, int n) {
  int i = blockIdx.x * blockDim.x + threadIdx.x;
  if (i < n) o[i] = a[i] + b[i];
}

__global__ void k_zero_u32(unsigned int* __restrict__ p, int n) {
  int i = blockIdx.x * blockDim.x + threadIdx.x;
  if (i < n) p[i] = 0u;
}

__device__ __forceinline__ float sigmoidf_(float x) { return 1.0f / (1.0f + expf(-x)); }

__global__ __launch_bounds__(128)
void k_lstm_cell(const u16* __restrict__ X, int fanin,          // [64, fanin] bf16
                 const u16* __restrict__ Wih,                    // [2048, fanin] bf16 row-major
                 const u16* __restrict__ Whh,                    // [2048, 512] bf16 row-major
                 const float* __restrict__ bias,                 // [2048] = bih + bhh
                 const u16* __restrict__ Hprev,                  // [64, 512] bf16 (t-1)
                 u16*  __restrict__ Hout_bf,                     // [64, 512] bf16 (t)
                 float* __restrict__ Hout_f32,                   // live h state (d_out)
                 float* __restrict__ Cstate,                     // live c state (d_out)
                 float* __restrict__ Yout)                       // outputs[t] or nullptr
{
  __shared__ float ldsg[4][16][16];

  const int tid  = threadIdx.x;
  const int lane = tid & 31;
  const int gate = tid >> 5;            // wave index = gate (0:i 1:f 2:g 3:o)
  const int lid  = lane & 15;
  const int half = lane >> 4;

  const int m0    = blockIdx.y * 16;    // batch-row tile
  const int n0    = blockIdx.x * 16;    // hidden-column tile within H
  const int nbase = gate * HID + n0;    // column tile within 4H (also W row base)

  v8f acc = {};

  // ---- gates += X @ Wih^T -------------------------------------------------
  {
    const u16* arow = X   + (size_t)(m0    + lid) * fanin;
    const u16* brow = Wih + (size_t)(nbase + lid) * fanin;
    for (int k0 = 0; k0 < fanin; k0 += 32) {
      // A 16x32 bf16, contiguous k order used by this code (B matches): lanes 0-15: K=[0..15]; lanes 16-31: K=[16..31]
      union { v16bf v; v8bf h[2]; } au;
      au.v = *(const v16bf*)(arow + k0 + half * 16);
      // B 32x16: lane n holds column n, K=k0..k0+15 (lanes 16-31: K=k0+16..k0+31)
      v16bf bb = *(const v16bf*)(brow + k0 + half * 16);
      acc = __builtin_amdgcn_wmma_f32_16x16x32_bf16(false, au.v, false, bb,
                                                    (short)0, acc, false, false);
    }
  }
  // ---- gates += Hprev @ Whh^T --------------------------------------------
  {
    const u16* arow = Hprev + (size_t)(m0    + lid) * HID;
    const u16* brow = Whh   + (size_t)(nbase + lid) * HID;
    for (int k0 = 0; k0 < HID; k0 += 32) {
      union { v16bf v; v8bf h[2]; } au;
      au.v = *(const v16bf*)(arow + k0 + half * 16);
      v16bf bb = *(const v16bf*)(brow + k0 + half * 16);
      acc = __builtin_amdgcn_wmma_f32_16x16x32_bf16(false, au.v, false, bb,
                                                    (short)0, acc, false, false);
    }
  }

  // ---- +bias, spill gate tile to LDS (C/D layout: vgpr r -> M=r+half*8) ---
  const float bv = bias[nbase + lid];
  #pragma unroll
  for (int r = 0; r < 8; ++r)
    ldsg[gate][r + half * 8][lid] = acc[r] + bv;

  __syncthreads();

  // ---- fused pointwise LSTM cell update for this 16x16 (batch, hidden) ----
  for (int e = tid; e < 256; e += 128) {
    const int m = e >> 4, n = e & 15;
    const int b = m0 + m;
    const int j = n0 + n;
    const float iv = ldsg[0][m][n];
    const float fv = ldsg[1][m][n];
    const float gv = ldsg[2][m][n];
    const float ov = ldsg[3][m][n];
    const size_t idx = (size_t)b * HID + j;
    const float c_old = Cstate[idx];
    const float c_new = sigmoidf_(fv) * c_old + sigmoidf_(iv) * tanhf(gv);
    const float h_new = sigmoidf_(ov) * tanhf(c_new);
    Cstate[idx]   = c_new;
    Hout_f32[idx] = h_new;
    Hout_bf[idx]  = f32_to_bf16_rne(h_new);
    if (Yout) Yout[idx] = h_new;
  }
}

// ---------------------------------------------------------------------------

extern "C" void kernel_launch(void* const* d_in, const int* in_sizes, int n_in,
                              void* d_out, int out_size, void* d_ws, size_t ws_size,
                              hipStream_t stream) {
  (void)in_sizes; (void)n_in; (void)out_size;

  const float* xs = (const float*)d_in[0];
  const float* Wih[NLAYER] = {(const float*)d_in[1], (const float*)d_in[5], (const float*)d_in[9]};
  const float* Whh[NLAYER] = {(const float*)d_in[2], (const float*)d_in[6], (const float*)d_in[10]};
  const float* bih[NLAYER] = {(const float*)d_in[3], (const float*)d_in[7], (const float*)d_in[11]};
  const float* bhh[NLAYER] = {(const float*)d_in[4], (const float*)d_in[8], (const float*)d_in[12]};

  float* outH = (float*)d_out;                              // [L,B,H]
  float* outC = outH + (size_t)NLAYER * BATCH * HID;        // [L,B,H]
  float* outY = outC + (size_t)NLAYER * BATCH * HID;        // [T,B,H]

  const int fanin[NLAYER] = {DIN, HID, HID};

  // ---- workspace carve-up (all bf16 regions 64B-aligned) ----
  char* ws = (char*)d_ws;
  size_t off = 0;
  float* biasF = (float*)(ws + off);
  off += (size_t)NLAYER * G4H * sizeof(float);
  off = (off + 63) & ~(size_t)63;

  u16* wihB[NLAYER]; u16* whhB[NLAYER];
  for (int l = 0; l < NLAYER; ++l) {
    wihB[l] = (u16*)(ws + off); off += (size_t)G4H * fanin[l] * sizeof(u16);
    off = (off + 63) & ~(size_t)63;
    whhB[l] = (u16*)(ws + off); off += (size_t)G4H * HID * sizeof(u16);
    off = (off + 63) & ~(size_t)63;
  }
  u16* xsB = (u16*)(ws + off);
  off += (size_t)T_STEPS * BATCH * DIN * sizeof(u16);
  off = (off + 63) & ~(size_t)63;
  u16* hB[NLAYER][2];
  for (int l = 0; l < NLAYER; ++l)
    for (int p = 0; p < 2; ++p) {
      hB[l][p] = (u16*)(ws + off);
      off += (size_t)BATCH * HID * sizeof(u16);
      off = (off + 63) & ~(size_t)63;
    }
  if (off > ws_size) return;  // workspace too small: nothing safe to do

  const int THR = 256;
  auto blocks = [](size_t n, int t) { return (unsigned)((n + t - 1) / t); };

  // ---- prep: bf16 weights + xs, fused biases, zeroed states ----
  for (int l = 0; l < NLAYER; ++l) {
    size_t nw = (size_t)G4H * fanin[l];
    k_cvt_bf16<<<blocks(nw, THR), THR, 0, stream>>>(Wih[l], wihB[l], (int)nw);
    size_t nh = (size_t)G4H * HID;
    k_cvt_bf16<<<blocks(nh, THR), THR, 0, stream>>>(Whh[l], whhB[l], (int)nh);
    k_add_f32<<<blocks(G4H, THR), THR, 0, stream>>>(bih[l], bhh[l], biasF + (size_t)l * G4H, G4H);
  }
  {
    size_t nx = (size_t)T_STEPS * BATCH * DIN;
    k_cvt_bf16<<<blocks(nx, THR), THR, 0, stream>>>(xs, xsB, (int)nx);
  }
  // zero bf16 h double-buffers and f32 h/c state regions in d_out
  for (int l = 0; l < NLAYER; ++l)
    for (int p = 0; p < 2; ++p)
      k_zero_u32<<<blocks((size_t)BATCH * HID / 2, THR), THR, 0, stream>>>(
          (unsigned int*)hB[l][p], BATCH * HID / 2);
  k_zero_u32<<<blocks((size_t)2 * NLAYER * BATCH * HID, THR), THR, 0, stream>>>(
      (unsigned int*)outH, 2 * NLAYER * BATCH * HID);

  // ---- the recurrence: one fused cell kernel per (t, layer) ----
  dim3 grid(HID / 16, BATCH / 16);   // (32, 4)
  for (int t = 0; t < T_STEPS; ++t) {
    const int wr = t & 1;            // write parity
    const int rd = 1 - wr;           // read parity (t-1); parity-1 buffer zeroed for t=0
    for (int l = 0; l < NLAYER; ++l) {
      const u16* X = (l == 0) ? (xsB + (size_t)t * BATCH * DIN) : hB[l - 1][wr];
      float* Yout = (l == NLAYER - 1) ? (outY + (size_t)t * BATCH * HID) : nullptr;
      k_lstm_cell<<<grid, 128, 0, stream>>>(
          X, fanin[l], wihB[l], whhB[l], biasF + (size_t)l * G4H,
          hB[l][rd], hB[l][wr],
          outH + (size_t)l * BATCH * HID,
          outC + (size_t)l * BATCH * HID,
          Yout);
    }
  }
}
